// Mamba2HSIClassifier_53618371723765
// MI455X (gfx1250) — matrix operand order repaired and checked by reading, not compiled, not run on hardware
//
#include <hip/hip_runtime.h>
#include <hip/hip_bf16.h>
#include <math.h>

// ---------------------------------------------------------------------------
// Mamba2 HSI classifier forward, MI455X (gfx1250), wave32 + WMMA f16->f32.
// All GEMM-shaped contractions run v_wmma_f32_16x16x32_f16 with f32 accum.
// Operands are pre-converted once to f16 (B operands pre-transposed to [N][K],
// zero-padded so K%32==0 and N%64==0), making every staging a contiguous row
// copy that goes through GLOBAL_LOAD_ASYNC_TO_LDS_B128 (ASYNCcnt),
// double-buffered against the WMMA loop.
// ---------------------------------------------------------------------------

typedef __attribute__((ext_vector_type(16))) _Float16 v16h;
typedef __attribute__((ext_vector_type(8)))  float    v8f;
typedef int v4i __attribute__((__vector_size__(16)));   // matches builtin sig

#define LDP 40   // LDS row stride in halves: 80B (16B aligned, conflict-free)

#if defined(__has_builtin)
#if __has_builtin(__builtin_amdgcn_global_load_async_to_lds_b128)
#define USE_ASYNC_LDS 1
#endif
#endif
#ifndef USE_ASYNC_LDS
#define USE_ASYNC_LDS 0
#endif

__device__ __forceinline__ void wait_async0() {
#if USE_ASYNC_LDS
#if __has_builtin(__builtin_amdgcn_s_wait_asynccnt)
  __builtin_amdgcn_s_wait_asynccnt(0);
#else
  asm volatile("s_wait_asynccnt 0x0" ::: "memory");
#endif
#endif
}

// Copy ROWS rows x 32 halves (64B) from global f16 (leading dim ldg halves)
// into LDS tile with row stride LDP.  16B chunks; ROWS*4 must be multiple of
// 256.  Async path: one GLOBAL_LOAD_ASYNC_TO_LDS_B128 per chunk per lane.
template <int ROWS>
__device__ __forceinline__ void stage_rows_f16(_Float16* dst, const _Float16* src,
                                               int ldg, int t) {
#pragma unroll
  for (int j = 0; j < (ROWS * 4) / 256; ++j) {
    const int id = j * 256 + t;
    const int r = id >> 2, c8 = (id & 3) * 8;
#if USE_ASYNC_LDS
    __builtin_amdgcn_global_load_async_to_lds_b128(
        (__attribute__((address_space(1))) v4i*)(src + (size_t)r * ldg + c8),
        (__attribute__((address_space(3))) v4i*)(dst + r * LDP + c8), 0, 0);
#else
    const uint4 v = *(const uint4*)(src + (size_t)r * ldg + c8);
    *(uint4*)(dst + r * LDP + c8) = v;
#endif
  }
}

// A fragment, 16x32 f16 (MxK): lane L holds row M=L&15; two 8-half runs at
// K = h*8 and K = 16+h*8, h=L>>4.
__device__ __forceinline__ v16h load_a_frag(const _Float16* base, int lane) {
  const int m = lane & 15, hh = lane >> 4;
  const _Float16* p0 = base + m * LDP + hh * 8;
  v16h r;
#pragma unroll
  for (int i = 0; i < 8; ++i) { r[i] = p0[i]; r[i + 8] = p0[16 + i]; }
  return r;
}

// B fragment, 32x16 f16 (KxN) staged as [N][K]: lane L holds column N=L&15,
// two 8-half runs at K = h*8 and K = 16+h*8, h=L>>4.
__device__ __forceinline__ v16h load_b_frag(const _Float16* base, int lane) {
  const int n = lane & 15, hh = lane >> 4;
  const _Float16* p = base + n * LDP + hh * 8;
  v16h r;
#pragma unroll
  for (int i = 0; i < 8; ++i) { r[i] = p[i]; r[i + 8] = p[16 + i]; }
  return r;
}

__device__ __forceinline__ v8f wmma_f16(v16h a, v16h b, v8f c) {
  return __builtin_amdgcn_wmma_f32_16x16x32_f16(false, a, false, b, (short)0, c,
                                                false, false);
}

__device__ __forceinline__ void wave_mma(const _Float16* la, const _Float16* lb,
                                         int wy, int wx, int lane, v8f acc[2][2]) {
  v16h af[2], bf[2];
#pragma unroll
  for (int mi = 0; mi < 2; ++mi) af[mi] = load_a_frag(la + (wy * 32 + mi * 16) * LDP, lane);
#pragma unroll
  for (int ni = 0; ni < 2; ++ni) bf[ni] = load_b_frag(lb + (wx * 32 + ni * 16) * LDP, lane);
#pragma unroll
  for (int mi = 0; mi < 2; ++mi)
#pragma unroll
    for (int ni = 0; ni < 2; ++ni) acc[mi][ni] = wmma_f16(af[mi], bf[ni], acc[mi][ni]);
}

__device__ __forceinline__ float sigmoidf_(float x) { return 1.f / (1.f + __expf(-x)); }

// ---------------------------------------------------------------------------
// f16 GEMM: C[z] = A[z] (MxK f16, lda) @ B[z]^T (B stored NxK f16, ldb) + bias.
// K%32==0 (zero-padded); B rows padded to N%64==0.  Block tile 128x64,
// 8 waves 4x2, wave 32x32 via 2x2 WMMA.  Double-buffered async staging.
// ---------------------------------------------------------------------------
template <bool OUT16>
__global__ __launch_bounds__(256) void k_gemm_h(
    const _Float16* __restrict__ A, int lda, unsigned long long sAz,
    const _Float16* __restrict__ B, int ldb, unsigned long long sBz,
    void* __restrict__ Cv, int ldc, unsigned long long sCz,
    const float* __restrict__ bias, int M, int N, int K) {
  __shared__ __align__(16) _Float16 ldsA[2][128 * LDP];
  __shared__ __align__(16) _Float16 ldsB[2][64 * LDP];
  const int t = threadIdx.x, lane = t & 31, w = t >> 5, wy = w >> 1, wx = w & 1;
  const int bm = blockIdx.y * 128, bn = blockIdx.x * 64;
  const _Float16* Az = A + (size_t)blockIdx.z * sAz + (size_t)bm * lda;
  const _Float16* Bz = B + (size_t)blockIdx.z * sBz + (size_t)bn * ldb;
  v8f acc[2][2] = {};
  const int nsteps = K >> 5;
  stage_rows_f16<128>(ldsA[0], Az, lda, t);
  stage_rows_f16<64>(ldsB[0], Bz, ldb, t);
  wait_async0();
  __syncthreads();
  for (int s = 0; s < nsteps; ++s) {
    const int cur = s & 1;
    if (s + 1 < nsteps) {  // prefetch next tiles while computing current
      stage_rows_f16<128>(ldsA[cur ^ 1], Az + (s + 1) * 32, lda, t);
      stage_rows_f16<64>(ldsB[cur ^ 1], Bz + (s + 1) * 32, ldb, t);
    }
    wave_mma(ldsA[cur], ldsB[cur], wy, wx, lane, acc);
    wait_async0();
    __syncthreads();
  }
  const int col16 = lane & 15, rb = (lane >> 4) * 8;
#pragma unroll
  for (int mi = 0; mi < 2; ++mi)
#pragma unroll
    for (int ni = 0; ni < 2; ++ni) {
      const int col = bn + wx * 32 + ni * 16 + col16;
      if (col >= N) continue;
      const float badd = bias ? bias[col] : 0.f;
#pragma unroll
      for (int r = 0; r < 8; ++r) {
        const int row = bm + wy * 32 + mi * 16 + rb + r;
        const size_t oi = (size_t)blockIdx.z * sCz + (size_t)row * ldc + col;
        if (OUT16) ((_Float16*)Cv)[oi] = (_Float16)(acc[mi][ni][r] + badd);
        else       ((float*)Cv)[oi]    = acc[mi][ni][r] + badd;
      }
    }
}

// ---------------------------------------------------------------------------
// f32 -> f16 convert with zero-padded leading dim: d[r*ldd+c] = s[r*C+c]
// ---------------------------------------------------------------------------
__global__ void k_cvt_pad(const float* __restrict__ s, _Float16* __restrict__ d,
                          int R, int C, int ldd) {
  const int idx = blockIdx.x * blockDim.x + threadIdx.x;
  if (idx >= R * ldd) return;
  const int r = idx / ldd, c = idx % ldd;
  d[idx] = (c < C) ? (_Float16)s[(size_t)r * C + c] : (_Float16)0.f;
}

// transpose+convert: d[c*ldd + r] = s[r*C + c] for c<C,r<R else 0; c in [0,CP)
__global__ void k_cvtT_pad(const float* __restrict__ s, _Float16* __restrict__ d,
                           int R, int C, int CP, int ldd) {
  const int idx = blockIdx.x * blockDim.x + threadIdx.x;
  if (idx >= CP * ldd) return;
  const int c = idx / ldd, r = idx % ldd;
  d[idx] = (c < C && r < R) ? (_Float16)s[(size_t)r * C + c] : (_Float16)0.f;
}

// ---------------------------------------------------------------------------
// dt = softplus(zx[:,1280:1288] + dt_bias)
// ---------------------------------------------------------------------------
__global__ void k_dt(const float* __restrict__ zx, const float* __restrict__ dt_bias,
                     float* __restrict__ dt) {
  const int idx = blockIdx.x * blockDim.x + threadIdx.x;
  if (idx >= 16384 * 8) return;
  const int h = idx & 7, row = idx >> 3;
  const float v = zx[(size_t)row * 1288 + 1280 + h] + dt_bias[h];
  dt[idx] = (v > 20.f) ? v : log1pf(__expf(v));
}

// ---------------------------------------------------------------------------
// Per (b,h,c): inclusive cumsum of A[h]*dt over the chunk; chunk decay.
// ---------------------------------------------------------------------------
__global__ __launch_bounds__(256) void k_cumsum(const float* __restrict__ dt,
                                                const float* __restrict__ A_log,
                                                float* __restrict__ Acs,
                                                float* __restrict__ cdecay) {
  const int z = blockIdx.x;
  const int c = z & 3, h = (z >> 2) & 7, b = z >> 5;
  const int q = threadIdx.x;
  __shared__ float s[256];
  const int row = b * 1024 + c * 256 + q;
  s[q] = -__expf(A_log[h]) * dt[row * 8 + h];
  __syncthreads();
  for (int off = 1; off < 256; off <<= 1) {
    const float v = (q >= off) ? s[q - off] : 0.f;
    __syncthreads();
    s[q] += v;
    __syncthreads();
  }
  Acs[(size_t)z * 256 + q] = s[q];
  if (q == 255) cdecay[z] = __expf(s[255]);
}

// ---------------------------------------------------------------------------
// Causal depthwise conv1d (taps=4) + SiLU; split into xh (f32 skip path),
// xdt^T (f16, [h*64+p][row] for SSD A/B staging), B (f32 + f16), C (f16).
// ---------------------------------------------------------------------------
__global__ void k_conv(const float* __restrict__ zx, const float* __restrict__ conv_w,
                       const float* __restrict__ conv_b, const float* __restrict__ dt,
                       float* __restrict__ xh, _Float16* __restrict__ xdtT,
                       float* __restrict__ Bb, _Float16* __restrict__ Bb16,
                       _Float16* __restrict__ Cb16) {
  const int idx = blockIdx.x * blockDim.x + threadIdx.x;
  if (idx >= 16384 * 768) return;
  const int ch = idx % 768, row = idx / 768;
  const int l = row & 1023;
  float acc = conv_b[ch];
#pragma unroll
  for (int tp = 0; tp < 4; ++tp) {
    const int lsrc = l - 3 + tp;
    if (lsrc >= 0)
      acc += conv_w[tp * 768 + ch] * zx[(size_t)(row - 3 + tp) * 1288 + 512 + ch];
  }
  const float s = acc * sigmoidf_(acc);
  if (ch < 512) {
    xh[(size_t)row * 512 + ch] = s;
    xdtT[(size_t)ch * 16384 + row] = (_Float16)(s * dt[row * 8 + (ch >> 6)]);
  } else if (ch < 640) {
    Bb[(size_t)row * 128 + (ch - 512)] = s;
    Bb16[(size_t)row * 128 + (ch - 512)] = (_Float16)s;
  } else {
    Cb16[(size_t)row * 128 + (ch - 640)] = (_Float16)s;
  }
}

// ---------------------------------------------------------------------------
// Yd: per (b,c,h): Y = (G .* exp(Acs_q - Acs_k)[k<=q]) @ X_h  (256x64, K=256)
// Fully branchless A staging: G and Acs both loaded unconditionally as
// 128-bit vectors; causal mask applied via select on the exp ARGUMENT
// (exp(-1e30)==0).  B operand async-staged from xdt^T in parallel.
// ---------------------------------------------------------------------------
__global__ __launch_bounds__(256) void k_yd(const float* __restrict__ G,
                                            const float* __restrict__ Acs,
                                            const _Float16* __restrict__ XdT,
                                            float* __restrict__ Y) {
  __shared__ __align__(16) _Float16 ldsA[128 * LDP];
  __shared__ __align__(16) _Float16 ldsB[64 * LDP];
  __shared__ __align__(16) float sAc[256];
  const int t = threadIdx.x, lane = t & 31, w = t >> 5, wy = w >> 1, wx = w & 1;
  const int z = blockIdx.z;
  const int h = z & 7, c = (z >> 3) & 3, b = z >> 5;
  const int bc = b * 4 + c, rowbase = bc * 256;
  const float* Gz = G + (size_t)bc * 65536;
  sAc[t] = Acs[((size_t)((b * 8 + h) * 4 + c)) * 256 + t];
  __syncthreads();
  const int bm = blockIdx.y * 128;
  const _Float16* Bsrc = XdT + (size_t)(h * 64) * 16384 + rowbase;
  v8f acc[2][2] = {};
  for (int k0 = 0; k0 < 256; k0 += 32) {
    stage_rows_f16<64>(ldsB, Bsrc + k0, 16384, t);  // DMA overlaps mask math
    {  // A: masked decayed G row, branchless (all loads unconditional)
      const int r = t >> 1, kof = (t & 1) * 16;
      const int q = bm + r;
      const float aq = sAc[q];
      const float4* gp = (const float4*)(Gz + q * 256 + k0 + kof);
      const float4* ap = (const float4*)(sAc + k0 + kof);
#pragma unroll
      for (int i4 = 0; i4 < 4; ++i4) {
        const float4 g4 = gp[i4];
        const float4 a4 = ap[i4];
        const float gv[4] = {g4.x, g4.y, g4.z, g4.w};
        const float av[4] = {a4.x, a4.y, a4.z, a4.w};
#pragma unroll
        for (int e = 0; e < 4; ++e) {
          const int kq = k0 + kof + i4 * 4 + e;
          const float d = (kq <= q) ? (aq - av[e]) : -1e30f;  // exp(-1e30)=0
          ldsA[r * LDP + kof + i4 * 4 + e] = (_Float16)(gv[e] * __expf(d));
        }
      }
    }
    wait_async0();
    __syncthreads();
    wave_mma(ldsA, ldsB, wy, wx, lane, acc);
    __syncthreads();
  }
  const int col16 = lane & 15, rb = (lane >> 4) * 8;
#pragma unroll
  for (int mi = 0; mi < 2; ++mi)
#pragma unroll
    for (int ni = 0; ni < 2; ++ni) {
      const int p = wx * 32 + ni * 16 + col16;
#pragma unroll
      for (int r = 0; r < 8; ++r) {
        const int q = bm + wy * 32 + mi * 16 + rb + r;
        Y[(size_t)(rowbase + q) * 512 + h * 64 + p] = acc[mi][ni][r];
      }
    }
}

// ---------------------------------------------------------------------------
// states: per (b,c,h): S[p][n] = sum_q X[q][p] * B[q][n] * exp(last-Acs[q])
// 64x128, K=256; waves 2x4.  A async from xdt^T; B decay-scaled sync staging.
// ---------------------------------------------------------------------------
__global__ __launch_bounds__(256) void k_states(const _Float16* __restrict__ XdT,
                                                const float* __restrict__ Bb,
                                                const float* __restrict__ Acs,
                                                float* __restrict__ states) {
  __shared__ __align__(16) _Float16 ldsA[64 * LDP];
  __shared__ __align__(16) _Float16 ldsB[128 * LDP];
  __shared__ __align__(16) float sAc[256];
  const int t = threadIdx.x, lane = t & 31, w = t >> 5, wy = w >> 2, wx = w & 3;
  const int z = blockIdx.z;
  const int h = z & 7, c = (z >> 3) & 3, b = z >> 5;
  const int rowbase = (b * 4 + c) * 256;
  sAc[t] = Acs[((size_t)((b * 8 + h) * 4 + c)) * 256 + t];
  __syncthreads();
  const float aLast = sAc[255];
  const _Float16* Asrc = XdT + (size_t)(h * 64) * 16384 + rowbase;
  v8f acc[2][2] = {};
  for (int k0 = 0; k0 < 256; k0 += 32) {
    stage_rows_f16<64>(ldsA, Asrc + k0, 16384, t);
    {  // B = Bb rows scaled by decay_states, staged [n][q]
      const int kk = t >> 3, n0 = (t & 7) * 16;
      const float ds = __expf(aLast - sAc[k0 + kk]);
      const int gq = rowbase + k0 + kk;
#pragma unroll
      for (int i = 0; i < 16; ++i)
        ldsB[(n0 + i) * LDP + kk] = (_Float16)(Bb[(size_t)gq * 128 + n0 + i] * ds);
    }
    wait_async0();
    __syncthreads();
    wave_mma(ldsA, ldsB, wy, wx, lane, acc);
    __syncthreads();
  }
  const int col16 = lane & 15, rb = (lane >> 4) * 8;
  float* Sz = states + (size_t)((b * 4 + c) * 8 + h) * 8192;
#pragma unroll
  for (int mi = 0; mi < 2; ++mi)
#pragma unroll
    for (int ni = 0; ni < 2; ++ni) {
      const int n = wx * 32 + ni * 16 + col16;
#pragma unroll
      for (int r = 0; r < 8; ++r) {
        const int p = wy * 32 + mi * 16 + rb + r;
        Sz[p * 128 + n] = acc[mi][ni][r];
      }
    }
}

// ---------------------------------------------------------------------------
// chunk recurrence over c=0..3; prev emitted as f16 (only feeds k_yo WMMA)
// ---------------------------------------------------------------------------
__global__ void k_scan(const float* __restrict__ states, const float* __restrict__ cdecay,
                       _Float16* __restrict__ prev) {
  const int idx = blockIdx.x * blockDim.x + threadIdx.x;
  if (idx >= (1 << 20)) return;
  const int n = idx & 127, p = (idx >> 7) & 63, h = (idx >> 13) & 7, b = idx >> 16;
  float carry = 0.f;
#pragma unroll
  for (int c = 0; c < 4; ++c) {
    const size_t off = ((size_t)((b * 4 + c) * 8 + h)) * 8192 + p * 128 + n;
    prev[off] = (_Float16)carry;
    carry = carry * cdecay[(b * 8 + h) * 4 + c] + states[off];
  }
}

// ---------------------------------------------------------------------------
// Yo: per (b,c,h): Y += exp(Acs[q])*(C @ prev^T) + D[h]*xh.  256x64, K=128.
// Both operands f16, async-staged.
// ---------------------------------------------------------------------------
__global__ __launch_bounds__(256) void k_yo(const _Float16* __restrict__ Cb16,
                                            const _Float16* __restrict__ prev,
                                            const float* __restrict__ Acs,
                                            const float* __restrict__ xh,
                                            const float* __restrict__ Dv,
                                            float* __restrict__ Y) {
  __shared__ __align__(16) _Float16 ldsA[128 * LDP];
  __shared__ __align__(16) _Float16 ldsB[64 * LDP];
  __shared__ __align__(16) float sAc[256];
  const int t = threadIdx.x, lane = t & 31, w = t >> 5, wy = w >> 1, wx = w & 1;
  const int z = blockIdx.z;
  const int h = z & 7, c = (z >> 3) & 3, b = z >> 5;
  const int rowbase = (b * 4 + c) * 256;
  sAc[t] = Acs[((size_t)((b * 8 + h) * 4 + c)) * 256 + t];
  const int bm = blockIdx.y * 128;
  const _Float16* Asrc = Cb16 + (size_t)(rowbase + bm) * 128;
  const _Float16* Bsrc = prev + (size_t)((b * 4 + c) * 8 + h) * 8192;
  __syncthreads();
  v8f acc[2][2] = {};
  for (int k0 = 0; k0 < 128; k0 += 32) {
    stage_rows_f16<128>(ldsA, Asrc + k0, 128, t);
    stage_rows_f16<64>(ldsB, Bsrc + k0, 128, t);
    wait_async0();
    __syncthreads();
    wave_mma(ldsA, ldsB, wy, wx, lane, acc);
    __syncthreads();
  }
  const int col16 = lane & 15, rb = (lane >> 4) * 8;
  const float dh = Dv[h];
#pragma unroll
  for (int mi = 0; mi < 2; ++mi)
#pragma unroll
    for (int ni = 0; ni < 2; ++ni) {
      const int p = wx * 32 + ni * 16 + col16;
#pragma unroll
      for (int r = 0; r < 8; ++r) {
        const int q = bm + wy * 32 + mi * 16 + rb + r;
        const size_t oi = (size_t)(rowbase + q) * 512 + h * 64 + p;
        Y[oi] += acc[mi][ni][r] * __expf(sAc[q]) + dh * xh[oi];
      }
    }
}

// ---------------------------------------------------------------------------
// gated RMSNorm -> f16 (feeds out-proj WMMA)
// ---------------------------------------------------------------------------
__global__ __launch_bounds__(256) void k_gate(const float* __restrict__ Y,
                                              const float* __restrict__ zx,
                                              const float* __restrict__ rms_w,
                                              _Float16* __restrict__ yg) {
  const int row = blockIdx.x, t = threadIdx.x;
  __shared__ float red[256];
  float yv[2];
  float ss = 0.f;
#pragma unroll
  for (int e = 0; e < 2; ++e) {
    const int ci = t + e * 256;
    float v = Y[(size_t)row * 512 + ci];
    const float zz = zx[(size_t)row * 1288 + ci];
    v *= zz * sigmoidf_(zz);
    yv[e] = v;
    ss += v * v;
  }
  red[t] = ss;
  __syncthreads();
  for (int s2 = 128; s2 > 0; s2 >>= 1) {
    if (t < s2) red[t] += red[t + s2];
    __syncthreads();
  }
  const float rr = rsqrtf(red[0] / 512.f + 1e-5f);
#pragma unroll
  for (int e = 0; e < 2; ++e) {
    const int ci = t + e * 256;
    yg[(size_t)row * 512 + ci] = (_Float16)(yv[e] * rr * rms_w[ci]);
  }
}

// ---------------------------------------------------------------------------
// LayerNorm -> f16 (classifier handled by one more WMMA GEMM, N=17 padded)
// ---------------------------------------------------------------------------
__global__ __launch_bounds__(256) void k_ln(const float* __restrict__ yo,
                                            const float* __restrict__ ln_g,
                                            const float* __restrict__ ln_b,
                                            _Float16* __restrict__ yn16) {
  const int row = blockIdx.x, t = threadIdx.x;
  __shared__ float red[256];
  const float v = yo[(size_t)row * 256 + t];
  red[t] = v;
  __syncthreads();
  for (int s2 = 128; s2 > 0; s2 >>= 1) {
    if (t < s2) red[t] += red[t + s2];
    __syncthreads();
  }
  const float mu = red[0] / 256.f;
  __syncthreads();
  const float d = v - mu;
  red[t] = d * d;
  __syncthreads();
  for (int s2 = 128; s2 > 0; s2 >>= 1) {
    if (t < s2) red[t] += red[t + s2];
    __syncthreads();
  }
  const float rstd = rsqrtf(red[0] / 256.f + 1e-5f);
  yn16[(size_t)row * 256 + t] = (_Float16)(d * rstd * ln_g[t] + ln_b[t]);
}

// ---------------------------------------------------------------------------
extern "C" void kernel_launch(void* const* d_in, const int* in_sizes, int n_in,
                              void* d_out, int out_size, void* d_ws, size_t ws_size,
                              hipStream_t stream) {
  const float* x       = (const float*)d_in[0];
  const float* w_inp   = (const float*)d_in[1];
  const float* b_inp   = (const float*)d_in[2];
  const float* w_in    = (const float*)d_in[3];
  const float* conv_w  = (const float*)d_in[4];
  const float* conv_b  = (const float*)d_in[5];
  const float* dt_bias = (const float*)d_in[6];
  const float* A_log   = (const float*)d_in[7];
  const float* Dv      = (const float*)d_in[8];
  const float* rms_w   = (const float*)d_in[9];
  const float* w_out   = (const float*)d_in[10];
  const float* ln_g    = (const float*)d_in[11];
  const float* ln_b    = (const float*)d_in[12];
  const float* w_cls   = (const float*)d_in[13];
  const float* b_cls   = (const float*)d_in[14];
  float* outp = (float*)d_out;

  char* p = (char*)d_ws;
  auto allocF = [&](size_t n) -> float* {
    float* r = (float*)p; p += (n * 4 + 255) & ~(size_t)255; return r;
  };
  auto allocH = [&](size_t n) -> _Float16* {
    _Float16* r = (_Float16*)p; p += (n * 2 + 255) & ~(size_t)255; return r;
  };
  _Float16* x16  = allocH((size_t)16384 * 224);  // x, K padded 200->224
  _Float16* wiT  = allocH((size_t)256 * 224);    // w_inp^T  [256][224]
  _Float16* winT = allocH((size_t)1344 * 256);   // w_in^T   [1344][256] (N pad)
  _Float16* woT  = allocH((size_t)256 * 512);    // w_out^T  [256][512]
  _Float16* wcT  = allocH((size_t)64 * 256);     // w_cls^T  [64][256]  (N pad)
  _Float16* u16  = allocH((size_t)16384 * 256);  // input-proj output (f16)
  float*    zx   = allocF((size_t)16384 * 1288); // in-proj output
  float*    dtb  = allocF((size_t)16384 * 8);
  float*    Acs  = allocF((size_t)512 * 256);
  float*    cdec = allocF((size_t)512);
  float*    xh   = allocF((size_t)16384 * 512);  // conv+silu x (skip path)
  _Float16* xdtT = allocH((size_t)512 * 16384);  // (x*dt)^T per channel
  float*    Bb   = allocF((size_t)16384 * 128);
  _Float16* Bb16 = allocH((size_t)16384 * 128);
  _Float16* Cb16 = allocH((size_t)16384 * 128);
  float*    G    = allocF((size_t)64 * 65536);
  float*    st   = allocF((size_t)512 * 8192);
  _Float16* pv16 = allocH((size_t)512 * 8192);
  float*    Y    = allocF((size_t)16384 * 512);
  _Float16* yg16 = allocH((size_t)16384 * 512);
  float*    yout = allocF((size_t)16384 * 256);
  _Float16* yn16 = allocH((size_t)16384 * 256);

  const dim3 blk(256, 1, 1);
  // 0) one-time f16 conversions / weight transposes (zero-padded)
  k_cvt_pad<<<dim3(14336, 1, 1), blk, 0, stream>>>(x, x16, 16384, 200, 224);
  k_cvtT_pad<<<dim3(224, 1, 1), blk, 0, stream>>>(w_inp, wiT, 200, 256, 256, 224);
  k_cvtT_pad<<<dim3(1344, 1, 1), blk, 0, stream>>>(w_in, winT, 256, 1288, 1344, 256);
  k_cvtT_pad<<<dim3(512, 1, 1), blk, 0, stream>>>(w_out, woT, 512, 256, 256, 512);
  k_cvtT_pad<<<dim3(64, 1, 1), blk, 0, stream>>>(w_cls, wcT, 256, 17, 64, 256);
  // 1) u16 = x @ w_inp + b_inp            (16384 x 256, K=224 padded)
  k_gemm_h<true><<<dim3(4, 128, 1), blk, 0, stream>>>(
      x16, 224, 0ull, wiT, 224, 0ull, u16, 256, 0ull, b_inp, 16384, 256, 224);
  // 2) zx = u @ w_in                      (16384 x 1288, K=256)
  k_gemm_h<false><<<dim3(21, 128, 1), blk, 0, stream>>>(
      u16, 256, 0ull, winT, 256, 0ull, zx, 1288, 0ull, nullptr, 16384, 1288, 256);
  // 3) dt softplus
  k_dt<<<dim3(512, 1, 1), blk, 0, stream>>>(zx, dt_bias, dtb);
  // 4) cumsum(A*dt) per chunk + chunk decay
  k_cumsum<<<dim3(512, 1, 1), blk, 0, stream>>>(dtb, A_log, Acs, cdec);
  // 5) conv + SiLU, split
  k_conv<<<dim3(49152, 1, 1), blk, 0, stream>>>(zx, conv_w, conv_b, dtb, xh, xdtT,
                                                Bb, Bb16, Cb16);
  // 6) G = C @ B^T per (b,c)              (256 x 256, K=128, Z=64)
  k_gemm_h<false><<<dim3(4, 2, 64), blk, 0, stream>>>(
      Cb16, 128, 32768ull, Bb16, 128, 32768ull, G, 256, 65536ull, nullptr, 256, 256, 128);
  // 7) Yd
  k_yd<<<dim3(1, 2, 512), blk, 0, stream>>>(G, Acs, xdtT, Y);
  // 8) states
  k_states<<<dim3(1, 1, 512), blk, 0, stream>>>(xdtT, Bb, Acs, st);
  // 9) inter-chunk recurrence
  k_scan<<<dim3(4096, 1, 1), blk, 0, stream>>>(st, cdec, pv16);
  // 10) Y += exp(Acs)*(C @ prev^T) + D*xh
  k_yo<<<dim3(1, 2, 512), blk, 0, stream>>>(Cb16, pv16, Acs, xh, Dv, Y);
  // 11) gated RMSNorm
  k_gate<<<dim3(16384, 1, 1), blk, 0, stream>>>(Y, zx, rms_w, yg16);
  // 12) out projection                    (16384 x 256, K=512)
  k_gemm_h<false><<<dim3(4, 128, 1), blk, 0, stream>>>(
      yg16, 512, 0ull, woT, 512, 0ull, yout, 256, 0ull, nullptr, 16384, 256, 512);
  // 13) LayerNorm
  k_ln<<<dim3(16384, 1, 1), blk, 0, stream>>>(yout, ln_g, ln_b, yn16);
  // 14) classifier                        (16384 x 17, K=256, N padded 64)
  k_gemm_h<false><<<dim3(1, 128, 1), blk, 0, stream>>>(
      yn16, 256, 0ull, wcT, 256, 0ull, outp, 17, 0ull, b_cls, 16384, 17, 256);
  (void)in_sizes; (void)n_in; (void)out_size; (void)ws_size;
}
